// DotProductAttention_26139170963678
// MI455X (gfx1250) — matrix operand order repaired and checked by reading, not compiled, not run on hardware
//
#include <hip/hip_runtime.h>

// ---------------------------------------------------------------------------
// Problem constants (from reference): B=2, S=2048, D=1024, H=16, HD=64
// ---------------------------------------------------------------------------
constexpr int cB  = 2;
constexpr int cS  = 2048;
constexpr int cD  = 1024;
constexpr int cH  = 16;
constexpr int cHD = 64;          // head dim
constexpr int cTOK = cB * cS;    // 4096 tokens
constexpr int cD3  = 3 * cD;     // 3072 qkv features

typedef __attribute__((ext_vector_type(16))) _Float16 v16h;
typedef __attribute__((ext_vector_type(8)))  _Float16 v8h;
typedef __attribute__((ext_vector_type(8)))  float    v8f;

union V16H { v16h v; v8h h[2]; };

// D = A(16x32 f16) * B(32x16 f16) + C(16x16 f32)
__device__ __forceinline__ v8f wmma_f16(v16h a, v16h b, v8f c) {
  return __builtin_amdgcn_wmma_f32_16x16x32_f16(
      /*neg_a=*/false, a, /*neg_b=*/false, b,
      /*c_mod=*/(short)0, c, /*reuse_a=*/false, /*reuse_b=*/false);
}

// A fragment (16x32 f16, row M = lane&15). Per ISA layout, lane group g=lane>>4
// holds K in [8g,8g+8) and [16+8g, 24+8g): two contiguous 16B runs.
__device__ __forceinline__ v16h load_a32(const _Float16* rowk, int g) {
  V16H u;
  u.h[0] = *(const v8h*)(rowk + 8 * g);
  u.h[1] = *(const v8h*)(rowk + 16 + 8 * g);
  return u.v;
}

// B fragment (32x16 f16, col N = lane&15, stored K-major per column):
// lane group g holds K in [8g,8g+8) and [16+8g, 24+8g): two contiguous 16B runs.
__device__ __forceinline__ v16h load_b32(const _Float16* colk, int g) {
  V16H u;
  u.h[0] = *(const v8h*)(colk + 8 * g);
  u.h[1] = *(const v8h*)(colk + 16 + 8 * g);
  return u.v;
}

__device__ __forceinline__ float rowmax16(float x) {
  #pragma unroll
  for (int off = 1; off < 16; off <<= 1) x = fmaxf(x, __shfl_xor(x, off, 32));
  return x;
}
__device__ __forceinline__ float rowsum16(float x) {
  #pragma unroll
  for (int off = 1; off < 16; off <<= 1) x += __shfl_xor(x, off, 32);
  return x;
}

// ---------------------------------------------------------------------------
// fp32 -> fp16 conversion
// ---------------------------------------------------------------------------
__global__ void conv_f32_f16_kernel(const float* __restrict__ src,
                                    _Float16* __restrict__ dst, int n) {
  int i = blockIdx.x * blockDim.x + threadIdx.x;
  if (i < n) dst[i] = (_Float16)src[i];
}

// ---------------------------------------------------------------------------
// QKV GEMM + bias + RoPE epilogue.
// Wave tile: 16 tokens x 64 features. K loop over D in steps of 32.
// Outputs: q (b,h,s,hd) pre-scaled by 1/sqrt(HD); k (b,h,s,hd); v^T (b,h,hd,s)
// ---------------------------------------------------------------------------
__global__ void qkv_rope_kernel(const _Float16* __restrict__ xh,
                                const _Float16* __restrict__ wh,
                                const float* __restrict__ bias,
                                const float* __restrict__ freqs,
                                _Float16* __restrict__ qbuf,
                                _Float16* __restrict__ kbuf,
                                _Float16* __restrict__ vT) {
  const int wid  = (blockIdx.x * blockDim.x + threadIdx.x) >> 5;
  const int lane = threadIdx.x & 31;
  const int g    = lane >> 4;
  const int hn   = lane & 15;

  const int NW  = cD3 / 64;            // 48 n-waves
  const int mt  = wid / NW;            // 16-token tile
  const int nw  = wid % NW;
  if (mt >= cTOK / 16) return;
  const int tok0 = mt * 16;
  const int nb   = nw * 64;

  v8f c[4] = {};
  const _Float16* arow = xh + (size_t)(tok0 + hn) * cD;

  for (int kk = 0; kk < cD; kk += 32) {
    v16h a = load_a32(arow + kk, g);
    #pragma unroll
    for (int t = 0; t < 4; ++t) {
      const _Float16* brow = wh + (size_t)(nb + t * 16 + hn) * cD + kk;
      v16h b = load_b32(brow, g);
      c[t] = wmma_f16(a, b, c[t]);
    }
  }

  const int blockq = nb / cD;          // 0=q 1=k 2=v (uniform per wave)
  const int head   = (nb % cD) / cHD;  // uniform per wave

  #pragma unroll
  for (int t = 0; t < 4; ++t) {
    const int hd = t * 16 + hn;        // within-head dim index
    const int n  = nb + t * 16 + hn;
    const float bn = bias[n];
    #pragma unroll
    for (int r = 0; r < 8; ++r) {
      const int tok = tok0 + r + 8 * g;
      const int bb  = tok / cS;
      const int ss  = tok % cS;
      float val = c[t][r] + bn;
      if (blockq < 2) {
        // RoPE: adjacent features live in adjacent lanes
        const int j = hd >> 1;
        const float f  = freqs[ss * (cHD / 2) + j];
        const float cs = cosf(f);
        const float sn = sinf(f);
        const float part = __shfl_xor(val, 1, 32);
        float out = (hd & 1) ? (part * sn + val * cs)
                             : (val * cs - part * sn);
        if (blockq == 0) out *= 0.125f;  // fold 1/sqrt(64) into q
        _Float16* dst = (blockq == 0) ? qbuf : kbuf;
        dst[((size_t)(bb * cH + head) * cS + ss) * cHD + hd] = (_Float16)out;
      } else {
        vT[((size_t)(bb * cH + head) * cHD + hd) * cS + ss] = (_Float16)val;
      }
    }
  }
}

// ---------------------------------------------------------------------------
// Flash attention: one wave per (b, h, 16-query tile); streams 32-key tiles.
// Causal mask computed analytically. Softmax online in f32.
// ---------------------------------------------------------------------------
__global__ void flash_attn_kernel(const _Float16* __restrict__ qbuf,
                                  const _Float16* __restrict__ kbuf,
                                  const _Float16* __restrict__ vT,
                                  _Float16* __restrict__ attnh) {
  __shared__ alignas(128) _Float16 lds_p[8][16 * 32];

  const int wslot = threadIdx.x >> 5;
  const int wid   = blockIdx.x * 8 + wslot;
  const int lane  = threadIdx.x & 31;
  const int g     = lane >> 4;
  const int hn    = lane & 15;

  const int qt = wid & 127;                 // S/16 = 128 query tiles
  const int h  = (wid >> 7) & 15;
  const int b  = wid >> 11;
  const int q0 = qt * 16;

  const _Float16* qrow  = qbuf + ((size_t)(b * cH + h) * cS + q0 + hn) * cHD;
  const _Float16* kbase = kbuf + (size_t)(b * cH + h) * cS * cHD;
  const _Float16* vbase = vT   + (size_t)(b * cH + h) * cHD * cS;

  const v16h aq0 = load_a32(qrow, g);       // hd 0..31
  const v16h aq1 = load_a32(qrow + 32, g);  // hd 32..63

  v8f o[4] = {};
  float mrun[8], lrun[8];
  #pragma unroll
  for (int r = 0; r < 8; ++r) { mrun[r] = -1e30f; lrun[r] = 0.0f; }

  _Float16* P = lds_p[wslot];
  const int kend = q0 + 16;

  for (int kt = 0; kt < kend; kt += 32) {
    if (kt + 32 < kend)
      __builtin_prefetch((const void*)(kbase + (size_t)(kt + 32 + hn) * cHD), 0, 1);

    const _Float16* krow0 = kbase + (size_t)(kt + hn) * cHD;
    const _Float16* krow1 = kbase + (size_t)(kt + 16 + hn) * cHD;

    v8f s0 = {}, s1 = {};
    s0 = wmma_f16(aq0, load_b32(krow0, g), s0);
    s0 = wmma_f16(aq1, load_b32(krow0 + 32, g), s0);
    s1 = wmma_f16(aq0, load_b32(krow1, g), s1);
    s1 = wmma_f16(aq1, load_b32(krow1 + 32, g), s1);

    float p0[8], p1[8];
    #pragma unroll
    for (int r = 0; r < 8; ++r) {
      const int qg  = q0 + r + 8 * g;
      const float v0 = (kt + hn      <= qg) ? s0[r] : -1e30f;
      const float v1 = (kt + 16 + hn <= qg) ? s1[r] : -1e30f;
      const float rm   = rowmax16(fmaxf(v0, v1));
      const float mnew = fmaxf(mrun[r], rm);
      const float corr = expf(mrun[r] - mnew);
      const float e0 = expf(v0 - mnew);
      const float e1 = expf(v1 - mnew);
      lrun[r] = lrun[r] * corr + rowsum16(e0 + e1);
      mrun[r] = mnew;
      p0[r] = e0; p1[r] = e1;
      #pragma unroll
      for (int t = 0; t < 4; ++t) o[t][r] *= corr;
    }

    // C-fragment -> A-fragment relayout of P via LDS (per-wave slot)
    #pragma unroll
    for (int r = 0; r < 8; ++r) {
      const int m = r + 8 * g;
      P[m * 32 + hn]      = (_Float16)p0[r];
      P[m * 32 + 16 + hn] = (_Float16)p1[r];
    }
    asm volatile("s_wait_dscnt 0" ::: "memory");
    V16H pa;
    pa.h[0] = *(const v8h*)(P + hn * 32 + 8 * g);
    pa.h[1] = *(const v8h*)(P + hn * 32 + 16 + 8 * g);
    asm volatile("s_wait_dscnt 0" ::: "memory");

    #pragma unroll
    for (int t = 0; t < 4; ++t) {
      const _Float16* vrow = vbase + (size_t)(t * 16 + hn) * cS + kt;
      o[t] = wmma_f16(pa.v, load_b32(vrow, g), o[t]);
    }
  }

  // normalize + store to (b, s, D) f16 for the projection GEMM
  #pragma unroll
  for (int r = 0; r < 8; ++r) {
    const int tok = b * cS + q0 + r + 8 * g;
    const float inv = 1.0f / lrun[r];
    #pragma unroll
    for (int t = 0; t < 4; ++t)
      attnh[(size_t)tok * cD + h * cHD + t * 16 + hn] =
          (_Float16)(o[t][r] * inv);
  }
}

// ---------------------------------------------------------------------------
// Output projection: out = attn @ w_proj^T + b_proj  (f32 result)
// ---------------------------------------------------------------------------
__global__ void proj_kernel(const _Float16* __restrict__ ah,
                            const _Float16* __restrict__ wh,
                            const float* __restrict__ bias,
                            float* __restrict__ out) {
  const int wid  = (blockIdx.x * blockDim.x + threadIdx.x) >> 5;
  const int lane = threadIdx.x & 31;
  const int g    = lane >> 4;
  const int hn   = lane & 15;

  const int NW = cD / 64;               // 16 n-waves
  const int mt = wid / NW;
  const int nw = wid % NW;
  if (mt >= cTOK / 16) return;
  const int tok0 = mt * 16;
  const int nb   = nw * 64;

  v8f c[4] = {};
  const _Float16* arow = ah + (size_t)(tok0 + hn) * cD;

  for (int kk = 0; kk < cD; kk += 32) {
    v16h a = load_a32(arow + kk, g);
    #pragma unroll
    for (int t = 0; t < 4; ++t) {
      const _Float16* brow = wh + (size_t)(nb + t * 16 + hn) * cD + kk;
      c[t] = wmma_f16(a, load_b32(brow, g), c[t]);
    }
  }

  #pragma unroll
  for (int t = 0; t < 4; ++t) {
    const int n = nb + t * 16 + hn;
    const float bn = bias[n];
    #pragma unroll
    for (int r = 0; r < 8; ++r) {
      const int tok = tok0 + r + 8 * g;
      out[(size_t)tok * cD + n] = c[t][r] + bn;
    }
  }
}

// ---------------------------------------------------------------------------
// Host-side launcher (graph-capture safe: only async launches on `stream`)
// ---------------------------------------------------------------------------
extern "C" void kernel_launch(void* const* d_in, const int* in_sizes, int n_in,
                              void* d_out, int out_size, void* d_ws, size_t ws_size,
                              hipStream_t stream) {
  const float* x      = (const float*)d_in[0];
  // d_in[1] = attn_mask (bool) — causal mask computed analytically, unused
  const float* freqs  = (const float*)d_in[2];
  const float* w_qkv  = (const float*)d_in[3];
  const float* b_qkv  = (const float*)d_in[4];
  const float* w_proj = (const float*)d_in[5];
  const float* b_proj = (const float*)d_in[6];
  float* out = (float*)d_out;

  // workspace layout (f16 buffers, all offsets 256B aligned)
  char* ws = (char*)d_ws;
  const size_t SZ_XH    = (size_t)cTOK * cD * 2;       //  8 MB
  const size_t SZ_WQKV  = (size_t)cD3 * cD * 2;        //  6 MB
  const size_t SZ_WPROJ = (size_t)cD * cD * 2;         //  2 MB
  const size_t SZ_HEADS = (size_t)cB * cH * cS * cHD * 2; // 8 MB each
  _Float16* xh     = (_Float16*)(ws);
  _Float16* wqkvh  = (_Float16*)(ws + SZ_XH);
  _Float16* wprojh = (_Float16*)(ws + SZ_XH + SZ_WQKV);
  _Float16* qbuf   = (_Float16*)(ws + SZ_XH + SZ_WQKV + SZ_WPROJ);
  _Float16* kbuf   = (_Float16*)(ws + SZ_XH + SZ_WQKV + SZ_WPROJ + SZ_HEADS);
  _Float16* vT     = (_Float16*)(ws + SZ_XH + SZ_WQKV + SZ_WPROJ + 2 * SZ_HEADS);
  _Float16* attnh  = (_Float16*)(ws + SZ_XH + SZ_WQKV + SZ_WPROJ + 3 * SZ_HEADS);

  // 1) convert inputs to f16
  {
    int n = cTOK * cD;
    conv_f32_f16_kernel<<<(n + 255) / 256, 256, 0, stream>>>(x, xh, n);
    n = cD3 * cD;
    conv_f32_f16_kernel<<<(n + 255) / 256, 256, 0, stream>>>(w_qkv, wqkvh, n);
    n = cD * cD;
    conv_f32_f16_kernel<<<(n + 255) / 256, 256, 0, stream>>>(w_proj, wprojh, n);
  }

  // 2) QKV GEMM + RoPE: (TOK/16)*(3D/64) = 256*48 = 12288 waves, 8 waves/block
  qkv_rope_kernel<<<12288 / 8, 256, 0, stream>>>(
      xh, wqkvh, b_qkv, freqs, qbuf, kbuf, vT);

  // 3) Flash attention: B*H*(S/16) = 4096 waves, 8 waves/block
  flash_attn_kernel<<<4096 / 8, 256, 0, stream>>>(qbuf, kbuf, vT, attnh);

  // 4) Projection: (TOK/16)*(D/64) = 256*16 = 4096 waves
  proj_kernel<<<4096 / 8, 256, 0, stream>>>(attnh, wprojh, b_proj, out);
}
